// Self_Attention_2d_48361331753478
// MI455X (gfx1250) — compile-verified, matrix operand order checked by reading, not run on hardware
//
#include <hip/hip_runtime.h>
#include <hip/hip_bf16.h>

typedef __attribute__((ext_vector_type(16))) __bf16 v16bf;
typedef __attribute__((ext_vector_type(8)))  float  v8f;

#define HEADS   12
#define RDIM    64
#define DDIM    768      // K of GEMM1
#define PDIM    4096     // M of both GEMMs, K of GEMM2
#define NBATCH  8
#define NCOLS2  (NBATCH * RDIM)   // 512 = N of GEMM2

// ---------------------------------------------------------------------------
// WMMA fragment helper (wave32, v_wmma_f32_16x16x32_bf16)
// A/B 16-bit fragment, ISA 7.12.2: lane<16 -> row=lane, K in {0..7, 16..23};
// lane>=16 -> row=lane-16, K in {8..15, 24..31}. Element e -> K = kb + (e<8?e:e+8).
// With a row-major LDS tile this lowers to two ds_load_b128 per fragment.
// ---------------------------------------------------------------------------
__device__ inline v16bf load_frag_rowmajor(const __bf16* tile, int ld) {
  int l  = threadIdx.x & 31;
  const __bf16* rp = tile + (size_t)(l & 15) * ld;
  int kb = (l < 16) ? 0 : 8;
  v16bf f;
#pragma unroll
  for (int e = 0; e < 16; ++e) {
    int k = kb + (e < 8 ? e : e + 8);
    f[e] = rp[k];
  }
  return f;
}

// D/C layout: VGPR j -> M = j + (lane<16 ? 0 : 8), N = lane & 15.
// wsb is [u][(n,r)][p] (p contiguous) -> each lane writes 8 contiguous bf16
// = one aligned 16-byte store.
__device__ inline void store_ws_frag(const v8f& acc, int q0, int nn, int p0,
                                     const float* __restrict__ vbias,
                                     __bf16* __restrict__ wsb) {
  int l  = threadIdx.x & 31;
  int q  = q0 + (l & 15);          // global output column = u*64 + r
  int mo = (l < 16) ? 0 : 8;
  int u  = q >> 6, r = q & 63;
  float bias = vbias[q];
  __bf16 buf[8];
#pragma unroll
  for (int j = 0; j < 8; ++j) buf[j] = (__bf16)(acc[j] + bias);
  *(uint4*)(wsb + ((size_t)(u * NCOLS2 + nn * RDIM + r)) * PDIM + p0 + mo) =
      *(const uint4*)buf;
}

__device__ inline void store_y_frag(const v8f& acc, int u, int c0, int p0,
                                    float* __restrict__ ybuf) {
  int l  = threadIdx.x & 31;
  int c  = c0 + (l & 15);          // global column = nidx*64 + r
  int mo = (l < 16) ? 0 : 8;
  int nidx = c >> 6, r = c & 63;
#pragma unroll
  for (int j = 0; j < 8; ++j) {
    int p = p0 + mo + j;
    ybuf[(((size_t)nidx * PDIM + p) * HEADS + u) * RDIM + r] = acc[j];
  }
}

// ---------------------------------------------------------------------------
// Kernel 0: f32 -> bf16 conversion (x and V)
// ---------------------------------------------------------------------------
__global__ void __launch_bounds__(256)
cvt_f32_bf16(const float* __restrict__ s, __bf16* __restrict__ d, int count) {
  int i = (blockIdx.x * 256 + threadIdx.x) * 4;
  if (i + 3 < count) {
    float4 v = *(const float4*)(s + i);
    d[i + 0] = (__bf16)v.x;
    d[i + 1] = (__bf16)v.y;
    d[i + 2] = (__bf16)v.z;
    d[i + 3] = (__bf16)v.w;
  }
}

// ---------------------------------------------------------------------------
// Kernel 1: ws = V.x^T + bias.  Per batch n: (P x D) @ (D x H*r).
// Block tile 128(M=p) x 128(N=u*64+r), 8 waves, wave tile 32x64
// (2 A frags x 4 B frags -> 8 WMMA per K-step of 32).
// Output bf16, TRANSPOSED layout [u][(n,r)][p] to feed GEMM2's B staging.
// ---------------------------------------------------------------------------
__global__ void __launch_bounds__(256)
gemm_ws(const __bf16* __restrict__ xb,    // [n][P][D]
        const __bf16* __restrict__ vb,    // [H*r][D]
        const float*  __restrict__ vbias, // [H*r]
        __bf16* __restrict__ wsb)         // [H][512][P]
{
  __shared__ __bf16 As[128 * 32];
  __shared__ __bf16 Bs[128 * 32];
  const int t    = threadIdx.x;
  const int wave = t >> 5;
  const int wm   = wave & 3;           // 4 M-subtiles of 32 -> 128
  const int wn   = wave >> 2;          // 2 N-subtiles of 64 -> 128
  const int bm   = blockIdx.x * 128;   // p base
  const int bq   = blockIdx.y * 128;   // output-column base (u*64+r)
  const int nn   = blockIdx.z;         // batch
  const __bf16* xrow = xb + (size_t)nn * PDIM * DDIM;

  v8f acc[2][4];
#pragma unroll
  for (int i = 0; i < 2; ++i)
#pragma unroll
    for (int s = 0; s < 4; ++s) acc[i][s] = (v8f){};

  const int srow = t >> 1;             // 0..127
  const int scol = (t & 1) * 16;

  for (int k0 = 0; k0 < DDIM; k0 += 32) {
    {   // stage A tile 128x32 (p x d) and B tile 128x32 (q x d), 16 bf16 each
      const uint4* sa = (const uint4*)(xrow + (size_t)(bm + srow) * DDIM + k0 + scol);
      uint4* da = (uint4*)(As + srow * 32 + scol);
      da[0] = sa[0]; da[1] = sa[1];
      const uint4* sb = (const uint4*)(vb + (size_t)(bq + srow) * DDIM + k0 + scol);
      uint4* db = (uint4*)(Bs + srow * 32 + scol);
      db[0] = sb[0]; db[1] = sb[1];
      if (k0 + 32 < DDIM) {   // hint next K tile (global_prefetch_b8)
        __builtin_prefetch(xrow + (size_t)(bm + srow) * DDIM + k0 + 32 + scol, 0, 1);
        __builtin_prefetch(vb   + (size_t)(bq + srow) * DDIM + k0 + 32 + scol, 0, 1);
      }
    }
    __syncthreads();
    v16bf a0 = load_frag_rowmajor(As + (wm * 32 + 0)  * 32, 32);
    v16bf a1 = load_frag_rowmajor(As + (wm * 32 + 16) * 32, 32);
    v16bf b[4];
#pragma unroll
    for (int s = 0; s < 4; ++s)
      b[s] = load_frag_rowmajor(Bs + (wn * 64 + s * 16) * 32, 32);
#pragma unroll
    for (int s = 0; s < 4; ++s) {
      acc[0][s] = __builtin_amdgcn_wmma_f32_16x16x32_bf16(false, a0, false, b[s], (short)0, acc[0][s], false, false);
      acc[1][s] = __builtin_amdgcn_wmma_f32_16x16x32_bf16(false, a1, false, b[s], (short)0, acc[1][s], false, false);
    }
    __syncthreads();
  }

#pragma unroll
  for (int i = 0; i < 2; ++i)
#pragma unroll
    for (int s = 0; s < 4; ++s)
      store_ws_frag(acc[i][s], bq + wn * 64 + s * 16, nn, bm + wm * 32 + i * 16,
                    vbias, wsb);
}

// ---------------------------------------------------------------------------
// Kernel 2: per head u, Y[p, c] = sum_j alpha_mat[u,p,j] * ws[u,c,j]
// alpha_mat is 2D-circulant: alpha_mat[u,(i,jj),(a,b)] = alpha[u,((a-i)&63)*64+((b-jj)&63)]
// A fragments are GENERATED from an 8KB bf16 copy of alpha[u] in LDS
// (never materialize the 805MB alpha_mat; A-operand HBM traffic ~= 0).
// Within a K-step of 32 the 'a' row index is constant (never crosses a 64 line).
// Block tile 128(M) x 128(N), wave tile 32x64 -> the 2 generated A fragments
// are amortized over 8 WMMAs. B tiles staged row-major [c][k] from the
// transposed wsb -> uint4 copies + b128 fragment loads.
// ---------------------------------------------------------------------------
__global__ void __launch_bounds__(256)
gemm_circ(const float*  __restrict__ alpha, // [H][P]
          const __bf16* __restrict__ wsb,   // [H][512][P]
          float* __restrict__ ybuf)         // [n][p][H][r]
{
  __shared__ __bf16 alph[PDIM];       // 8 KB: whole per-head alpha
  __shared__ __bf16 Bs[128 * 32];     // [c][k]
  const int t  = threadIdx.x;
  const int u  = blockIdx.z;
  const int bm = blockIdx.x * 128;
  const int bn = blockIdx.y * 128;    // column base within 512

  const float* ap = alpha + (size_t)u * PDIM;
  for (int i = t; i < PDIM; i += 256) alph[i] = (__bf16)ap[i];

  const int wave = t >> 5, wm = wave & 3, wn = wave >> 2, l = t & 31;
  const int lm = l & 15;
  const int kb = (l < 16) ? 0 : 8;
  const int p0 = bm + wm * 32 + lm;          // fragment-0 row for this lane
  const int i0 = p0 >> 6,        j0 = p0 & 63;
  const int i1 = (p0 + 16) >> 6, j1 = (p0 + 16) & 63;

  const __bf16* wsu = wsb + (size_t)u * NCOLS2 * PDIM;

  v8f acc[2][4];
#pragma unroll
  for (int i = 0; i < 2; ++i)
#pragma unroll
    for (int s = 0; s < 4; ++s) acc[i][s] = (v8f){};

  const int srow = t >> 1;             // 0..127 (c local)
  const int scol = (t & 1) * 16;
  __syncthreads();                     // alph ready

  for (int k0 = 0; k0 < PDIM; k0 += 32) {
    {   // stage B tile [c][k]: p contiguous in the transposed wsb
      const uint4* sb = (const uint4*)(wsu + (size_t)(bn + srow) * PDIM + k0 + scol);
      uint4* db = (uint4*)(Bs + srow * 32 + scol);
      db[0] = sb[0]; db[1] = sb[1];
      if (k0 + 32 < PDIM)
        __builtin_prefetch(wsu + (size_t)(bn + srow) * PDIM + k0 + 32 + scol, 0, 1);
    }
    __syncthreads();

    // generate circulant A fragments from LDS alpha
    const int abase = k0 >> 6;       // 'a' is constant within this K-step
    const int bbase = k0 & 32;
    const int ar0 = ((abase - i0) & 63) * 64;
    const int ar1 = ((abase - i1) & 63) * 64;
    v16bf a0, a1;
#pragma unroll
    for (int e = 0; e < 16; ++e) {
      int K = kb + (e < 8 ? e : e + 8);
      a0[e] = alph[ar0 + ((bbase + K - j0) & 63)];
      a1[e] = alph[ar1 + ((bbase + K - j1) & 63)];
    }
    v16bf b[4];
#pragma unroll
    for (int s = 0; s < 4; ++s)
      b[s] = load_frag_rowmajor(Bs + (wn * 64 + s * 16) * 32, 32);
#pragma unroll
    for (int s = 0; s < 4; ++s) {
      acc[0][s] = __builtin_amdgcn_wmma_f32_16x16x32_bf16(false, a0, false, b[s], (short)0, acc[0][s], false, false);
      acc[1][s] = __builtin_amdgcn_wmma_f32_16x16x32_bf16(false, a1, false, b[s], (short)0, acc[1][s], false, false);
    }
    __syncthreads();
  }

#pragma unroll
  for (int i = 0; i < 2; ++i)
#pragma unroll
    for (int s = 0; s < 4; ++s)
      store_y_frag(acc[i][s], u, bn + wn * 64 + s * 16, bm + wm * 32 + i * 16, ybuf);
}

// ---------------------------------------------------------------------------
// Kernel 3: out[n,p,v*64+r] = b_mix[v] + sum_u W_mix[u,v] * y[n,p,u,r]
// One thread per (n,p,r); fully coalesced along r; W/b in LDS.
// ---------------------------------------------------------------------------
__global__ void __launch_bounds__(256)
headmix(const float* __restrict__ ybuf,  // [n][p][H][r]
        const float* __restrict__ W,     // [H][H]
        const float* __restrict__ bmix,  // [H]
        float* __restrict__ out)         // [n][p][H*r]
{
  __shared__ float Ws[HEADS * HEADS];
  __shared__ float bsh[HEADS];
  int t = threadIdx.x;
  if (t < HEADS * HEADS) Ws[t] = W[t];
  if (t < HEADS)         bsh[t] = bmix[t];
  __syncthreads();

  size_t idx = (size_t)blockIdx.x * 256 + t;   // n*P*r = 2,097,152 exactly
  int    r   = (int)(idx & 63);
  size_t np  = idx >> 6;                       // n*P + p
  const float* yp = ybuf + (np * HEADS) * RDIM + r;
  float yv[HEADS];
#pragma unroll
  for (int u = 0; u < HEADS; ++u) yv[u] = yp[(size_t)u * RDIM];
  float* op = out + np * (HEADS * RDIM) + r;
#pragma unroll
  for (int v = 0; v < HEADS; ++v) {
    float s = bsh[v];
#pragma unroll
    for (int u = 0; u < HEADS; ++u) s += yv[u] * Ws[u * HEADS + v];
    op[(size_t)v * RDIM] = s;
  }
}

// ---------------------------------------------------------------------------
// Launch
// ---------------------------------------------------------------------------
extern "C" void kernel_launch(void* const* d_in, const int* in_sizes, int n_in,
                              void* d_out, int out_size, void* d_ws, size_t ws_size,
                              hipStream_t stream) {
  (void)in_sizes; (void)n_in; (void)out_size; (void)ws_size;

  const float* x     = (const float*)d_in[0];  // (8,4096,768)
  const float* alpha = (const float*)d_in[1];  // (12,4096)
  const float* V     = (const float*)d_in[2];  // (12,64,768)
  const float* vbias = (const float*)d_in[3];  // (12,64,1) -> flat 768
  const float* Wmix  = (const float*)d_in[4];  // (12,12)
  const float* bmix  = (const float*)d_in[5];  // (12,)
  float* out = (float*)d_out;                  // (8,4096,768)

  // workspace partition (all offsets 256B aligned)
  char* ws = (char*)d_ws;
  const size_t n_xb  = (size_t)NBATCH * PDIM * DDIM;       // 25,165,824 bf16
  const size_t n_vb  = (size_t)HEADS * RDIM * DDIM;        //    589,824 bf16
  const size_t n_wsb = (size_t)HEADS * NCOLS2 * PDIM;      // 25,165,824 bf16
  __bf16* xb   = (__bf16*)(ws);
  __bf16* vb   = (__bf16*)(ws + n_xb * 2);
  __bf16* wsb  = (__bf16*)(ws + n_xb * 2 + 1179648);
  float*  ybuf = (float*) (ws + n_xb * 2 + 1179648 + n_wsb * 2);

  // 0) f32 -> bf16
  cvt_f32_bf16<<<(int)(n_xb / 4 / 256), 256, 0, stream>>>(x, xb, (int)n_xb);
  cvt_f32_bf16<<<(int)(n_vb / 4 / 256), 256, 0, stream>>>(V, vb, (int)n_vb);

  // 1) ws GEMM: grid = (M tiles 4096/128, N tiles 768/128, batch)
  gemm_ws<<<dim3(PDIM / 128, (HEADS * RDIM) / 128, NBATCH), 256, 0, stream>>>(xb, vb, vbias, wsb);

  // 2) circulant GEMM: grid = (M tiles 4096/128, N tiles 512/128, heads)
  gemm_circ<<<dim3(PDIM / 128, NCOLS2 / 128, HEADS), 256, 0, stream>>>(alpha, wsb, ybuf);

  // 3) head mix: n*P*r threads
  headmix<<<(NBATCH * PDIM * RDIM) / 256, 256, 0, stream>>>(ybuf, Wmix, bmix, out);
}
